// Emotion_diarization_single_ContentCrossAttetion_22969485099322
// MI455X (gfx1250) — compile-verified, matrix operand order checked by reading, not run on hardware
//
#include <hip/hip_runtime.h>

typedef __bf16 bf16;
typedef __bf16 v16bf __attribute__((ext_vector_type(16)));
typedef __bf16 v8bf  __attribute__((ext_vector_type(8)));
typedef float  v8f   __attribute__((ext_vector_type(8)));

// pointee/addrspace types for the async-DMA builtin (param type taken from the
// compiler diagnostic: 'int __vector(4) __device__ *' / '... __shared__ *')
typedef int i32x4 __attribute__((vector_size(16)));
typedef __attribute__((address_space(1))) i32x4 as1_i32x4;
typedef __attribute__((address_space(3))) i32x4 as3_i32x4;

#define NB   8
#define ND   256
#define NTC  2048
#define NTE  2048
#define NTGT 256
#define INV_SCALE 0.0625f   // 1/sqrt(256)

#if __has_builtin(__builtin_amdgcn_global_load_async_to_lds_b128)
#define HAS_ASYNC 1
#else
#define HAS_ASYNC 0
#endif

#if HAS_ASYNC
#if __has_builtin(__builtin_amdgcn_s_wait_asynccnt)
#define WAIT_ASYNC() __builtin_amdgcn_s_wait_asynccnt(0)
#else
#define WAIT_ASYNC() asm volatile("s_wait_asynccnt 0x0" ::: "memory")
#endif
#else
#define WAIT_ASYNC()
#endif

// ---------- small helpers ----------
__device__ __forceinline__ v8f v8f_splat(float v) {
  v8f r;
#pragma unroll
  for (int i = 0; i < 8; ++i) r[i] = v;
  return r;
}
__device__ __forceinline__ v8bf v8bf_zero() {
  v8bf r;
#pragma unroll
  for (int i = 0; i < 8; ++i) r[i] = (bf16)0.0f;
  return r;
}
__device__ __forceinline__ v16bf cat8(v8bf lo, v8bf hi) {
  return __builtin_shufflevector(lo, hi, 0,1,2,3,4,5,6,7,8,9,10,11,12,13,14,15);
}
// loads the two 8-element K-chunks a lane owns: [0,8) at p and [16,24) at p
__device__ __forceinline__ v16bf load_frag(const bf16* p) {
  v8bf lo = *(const v8bf*)p;
  v8bf hi = *(const v8bf*)(p + 16);
  return cat8(lo, hi);
}
__device__ __forceinline__ v8f wmma_bf16(v16bf a, v16bf b, v8f c) {
  return __builtin_amdgcn_wmma_f32_16x16x32_bf16(false, a, false, b, (short)0, c,
                                                 false, false);
}
// 16-byte global -> LDS copy (async DMA when available)
__device__ __forceinline__ void copy16(const bf16* g, bf16* l) {
#if HAS_ASYNC
  __builtin_amdgcn_global_load_async_to_lds_b128(
      (as1_i32x4*)const_cast<bf16*>(g), (as3_i32x4*)l, 0, 0);
#else
  *(v8bf*)l = *(const v8bf*)g;
#endif
}

// ---------- pack kernels ----------
// src: [B][256][T] f32  ->  dst: [B][T][dst_stride] bf16 (first 256 cols)
__global__ __launch_bounds__(256) void k_transpose_pack(const float* __restrict__ src,
                                                        bf16* __restrict__ dst,
                                                        int T, int dst_stride) {
  __shared__ float tile[32][33];
  const int b  = blockIdx.z;
  const int t0 = blockIdx.x * 32;
  const int d0 = blockIdx.y * 32;
  const int tx = threadIdx.x, ty = threadIdx.y;
  const float* s = src + ((size_t)b * ND + d0) * T + t0;
#pragma unroll
  for (int r = 0; r < 32; r += 8) tile[r + ty][tx] = s[(size_t)(r + ty) * T + tx];
  __syncthreads();
  bf16* dp = dst + ((size_t)b * T + t0) * dst_stride + d0;
#pragma unroll
  for (int r = 0; r < 32; r += 8)
    dp[(size_t)(r + ty) * dst_stride + tx] = (bf16)tile[tx][r + ty];
}

__global__ __launch_bounds__(256) void k_convert_flat(const float* __restrict__ src,
                                                      bf16* __restrict__ dst, int n) {
  int i = blockIdx.x * 256 + threadIdx.x;
  if (i < n) dst[i] = (bf16)src[i];
}

// w: [256 o][512 i][3 k] f32  ->  wt: [3 k][256 o][512 i] bf16
__global__ __launch_bounds__(256) void k_pack_w(const float* __restrict__ w,
                                                bf16* __restrict__ wt) {
  int idx = blockIdx.x * 256 + threadIdx.x;     // over 3*256*512
  if (idx >= 3 * 256 * 512) return;
  int i = idx % 512;
  int o = (idx / 512) % 256;
  int k = idx / (512 * 256);
  wt[idx] = (bf16)w[((size_t)o * 512 + i) * 3 + k];
}

// ---------- fused cross-attention (flash style, async double-buffered) ----------
// x:   [B][TC][512] bf16, cols [0,256) = content rows (A operand), cols [256,512) written
// ebt: [B][TE][256] bf16  (E row-major -> B-operand for S = C * E^T)
// ebn: [B][256][TE] bf16  (E^T -> B-operand for P * E)
__global__ __launch_bounds__(256) void k_attn(const bf16* __restrict__ x,
                                              bf16* __restrict__ xw,
                                              const bf16* __restrict__ ebt,
                                              const bf16* __restrict__ ebn) {
  __shared__ __align__(16) bf16 et_buf[2][32][256];   // E rows   (S B-frags)
  __shared__ __align__(16) bf16 en_buf[2][256][32];   // E^T rows (PV B-frags)
  __shared__ __align__(16) bf16 p_tile[8][16][32];    // per-wave P staging

  const int tid  = threadIdx.x;
  const int lane = tid & 31;
  const int wave = tid >> 5;
  const int colj = lane & 15;   // M (A rows) or N (B cols)
  const int half = lane >> 4;   // K chunk / row-half select
  const int b  = blockIdx.y;
  const int t0 = blockIdx.x * 128 + wave * 16;

  const bf16* ebt_b = ebt + (size_t)b * NTE * ND;
  const bf16* ebn_b = ebn + (size_t)b * ND * NTE;

  // block-cooperative async stage of one 32-column chunk into buffer `buf`
  auto stage = [&](int s0, int buf) {
#pragma unroll
    for (int it = 0; it < 4; ++it) {          // 32 rows x 512B, contiguous rows
      int q = it * 256 + tid;
      int sr = q >> 5, doff = (q & 31) * 8;
      copy16(ebt_b + (size_t)(s0 + sr) * ND + doff, &et_buf[buf][sr][doff]);
    }
#pragma unroll
    for (int it = 0; it < 4; ++it) {          // 256 rows x 64B
      int q = it * 256 + tid;
      int d = q >> 2, soff = (q & 3) * 8;
      copy16(ebn_b + (size_t)d * NTE + s0 + soff, &en_buf[buf][d][soff]);
    }
  };

  // preload the 16x256 content A-fragments (8 K-steps)
  const bf16* crow = x + ((size_t)b * NTC + (t0 + colj)) * 512 + half * 8;
  v16bf af[8];
#pragma unroll
  for (int ks = 0; ks < 8; ++ks) af[ks] = load_frag(crow + ks * 32);

  v8f acc[16];
#pragma unroll
  for (int nt = 0; nt < 16; ++nt) acc[nt] = v8f_splat(0.0f);
  float mi[8], li[8];
#pragma unroll
  for (int r = 0; r < 8; ++r) { mi[r] = -1.0e30f; li[r] = 0.0f; }

  stage(0, 0);
  WAIT_ASYNC();
  __syncthreads();
  int cur = 0;

  for (int s0 = 0; s0 < NTE; s0 += 32) {
    if (s0 + 32 < NTE) stage(s0 + 32, cur ^ 1);   // overlap DMA with compute

    // ---- S = (C * E^T) * inv_scale : two 16x16 tiles, B-frags from LDS ----
    v8f st[2];
#pragma unroll
    for (int j = 0; j < 2; ++j) {
      v8f sj = v8f_splat(0.0f);
      const bf16* erow = &et_buf[cur][j * 16 + colj][half * 8];
#pragma unroll
      for (int ks = 0; ks < 8; ++ks) sj = wmma_bf16(af[ks], load_frag(erow + ks * 32), sj);
      st[j] = sj;
    }

    // ---- online softmax over the 32 new columns ----
    float fac[8];
#pragma unroll
    for (int r = 0; r < 8; ++r) {
      float a0 = st[0][r] * INV_SCALE;
      float a1 = st[1][r] * INV_SCALE;
      float mx = fmaxf(a0, a1);
#pragma unroll
      for (int off = 1; off < 16; off <<= 1) mx = fmaxf(mx, __shfl_xor(mx, off, 32));
      float mn = fmaxf(mi[r], mx);
      fac[r] = __expf(mi[r] - mn);
      mi[r] = mn;
      float p0 = __expf(a0 - mn);
      float p1 = __expf(a1 - mn);
      float rs = p0 + p1;
#pragma unroll
      for (int off = 1; off < 16; off <<= 1) rs += __shfl_xor(rs, off, 32);
      li[r] = li[r] * fac[r] + rs;
      // stash P in D-layout: row = 8*half + r, col = 16*j + colj
      p_tile[wave][8 * half + r][colj]      = (bf16)p0;
      p_tile[wave][8 * half + r][16 + colj] = (bf16)p1;
    }

    // rescale running accumulators
#pragma unroll
    for (int nt = 0; nt < 16; ++nt)
#pragma unroll
      for (int r = 0; r < 8; ++r) acc[nt][r] *= fac[r];

    // re-read P as an A-fragment (lane = row, halves = K chunks)
    v16bf pf = load_frag(&p_tile[wave][colj][half * 8]);

    // ---- acc += P * E, B-frags from LDS ----
#pragma unroll
    for (int nt = 0; nt < 16; ++nt)
      acc[nt] = wmma_bf16(pf, load_frag(&en_buf[cur][nt * 16 + colj][half * 8]), acc[nt]);

    WAIT_ASYNC();        // next chunk's DMA done
    __syncthreads();     // everyone finished reading `cur`, next buffer filled
    cur ^= 1;
  }

  // ---- normalize and write weighted_emo into x[:, :, 256:512] ----
  float inv[8];
#pragma unroll
  for (int r = 0; r < 8; ++r) inv[r] = 1.0f / li[r];
#pragma unroll
  for (int nt = 0; nt < 16; ++nt)
#pragma unroll
    for (int r = 0; r < 8; ++r) {
      int t = t0 + 8 * half + r;
      xw[((size_t)b * NTC + t) * 512 + 256 + nt * 16 + colj] = (bf16)(acc[nt][r] * inv[r]);
    }
}

// ---------- conv1d(512->256, k=3, pad=1) as WMMA GEMM, async weight staging ----------
__global__ __launch_bounds__(256) void k_conv(const bf16* __restrict__ x,
                                              const bf16* __restrict__ wt,
                                              const float* __restrict__ bias,
                                              float* __restrict__ out) {
  __shared__ __align__(16) bf16 w_buf[2][256][32];   // weight tile per (k, i-chunk)

  const int tid  = threadIdx.x;
  const int lane = tid & 31;
  const int wave = tid >> 5;
  const int colj = lane & 15;
  const int half = lane >> 4;
  const int b  = blockIdx.y;
  const int t0 = blockIdx.x * 128 + wave * 16;

  auto stagew = [&](int step, int buf) {
    int k = step >> 4, ich = step & 15;
#pragma unroll
    for (int it = 0; it < 4; ++it) {        // 256 rows x 64B
      int q = it * 256 + tid;
      int o = q >> 2, ioff = (q & 3) * 8;
      copy16(wt + ((size_t)(k * 256 + o)) * 512 + ich * 32 + ioff, &w_buf[buf][o][ioff]);
    }
  };

  v8f acc[16];
#pragma unroll
  for (int nt = 0; nt < 16; ++nt) acc[nt] = v8f_splat(bias[nt * 16 + colj]);

  stagew(0, 0);
  WAIT_ASYNC();
  __syncthreads();
  int cur = 0;

  for (int step = 0; step < 48; ++step) {   // step = k*16 + ich
    if (step + 1 < 48) stagew(step + 1, cur ^ 1);

    const int k = step >> 4, ich = step & 15;
    const int t = t0 + colj + k - 1;
    const bool valid = (t >= 0) && (t < NTC);
    const bf16* xr = x + ((size_t)b * NTC + (valid ? t : 0)) * 512 + half * 8;
    v8bf lo = v8bf_zero(), hi = v8bf_zero();
    if (valid) {
      lo = *(const v8bf*)(xr + ich * 32);
      hi = *(const v8bf*)(xr + ich * 32 + 16);
    }
    v16bf afr = cat8(lo, hi);

#pragma unroll
    for (int nt = 0; nt < 16; ++nt)
      acc[nt] = wmma_bf16(afr, load_frag(&w_buf[cur][nt * 16 + colj][half * 8]), acc[nt]);

    WAIT_ASYNC();
    __syncthreads();
    cur ^= 1;
  }

#pragma unroll
  for (int nt = 0; nt < 16; ++nt)
#pragma unroll
    for (int r = 0; r < 8; ++r) {
      int t = t0 + 8 * half + r;
      int o = nt * 16 + colj;
      out[((size_t)b * NTGT + o) * NTC + t] = acc[nt][r];
    }
}

// ---------- launcher ----------
extern "C" void kernel_launch(void* const* d_in, const int* in_sizes, int n_in,
                              void* d_out, int out_size, void* d_ws, size_t ws_size,
                              hipStream_t stream) {
  (void)in_sizes; (void)n_in; (void)out_size; (void)ws_size;
  const float* content = (const float*)d_in[0];   // [8][256][2048]
  const float* emotion = (const float*)d_in[1];   // [8][256][2048]
  const float* conv_w  = (const float*)d_in[2];   // [256][512][3]
  const float* conv_b  = (const float*)d_in[3];   // [256]
  float* out = (float*)d_out;                     // [8][256][2048]

  char* w = (char*)d_ws;
  const size_t X_BYTES   = (size_t)NB * NTC * 512 * sizeof(bf16);   // 16 MiB
  const size_t EBT_BYTES = (size_t)NB * NTE * ND  * sizeof(bf16);   //  8 MiB
  const size_t EBN_BYTES = EBT_BYTES;
  bf16* x   = (bf16*)w;
  bf16* ebt = (bf16*)(w + X_BYTES);
  bf16* ebn = (bf16*)(w + X_BYTES + EBT_BYTES);
  bf16* wt  = (bf16*)(w + X_BYTES + EBT_BYTES + EBN_BYTES);

  dim3 tp_grid(NTC / 32, ND / 32, NB), tp_blk(32, 8);
  k_transpose_pack<<<tp_grid, tp_blk, 0, stream>>>(content, x, NTC, 512);
  k_transpose_pack<<<tp_grid, tp_blk, 0, stream>>>(emotion, ebt, NTE, ND);

  const int nflat = NB * ND * NTE;
  k_convert_flat<<<(nflat + 255) / 256, 256, 0, stream>>>(emotion, ebn, nflat);
  k_pack_w<<<(3 * 256 * 512) / 256, 256, 0, stream>>>(conv_w, wt);

  k_attn<<<dim3(NTC / 128, NB), 256, 0, stream>>>(x, x, ebt, ebn);
  k_conv<<<dim3(NTC / 128, NB), 256, 0, stream>>>(x, wt, conv_b, out);
}
